// PointCloudFiLM_1554778161347
// MI455X (gfx1250) — matrix operand order repaired and checked by reading, not compiled, not run on hardware
//
#include <hip/hip_runtime.h>
#include <hip/hip_bf16.h>

typedef __attribute__((ext_vector_type(16))) _Float16 v16h;
typedef __attribute__((ext_vector_type(8)))  _Float16 v8h;
typedef __attribute__((ext_vector_type(8)))  float    v8f;

#define B_    128
#define N_    8192
#define C1_   64
#define C2_   128
#define C3_   256
#define IMG_  512
#define EPS_  1e-5f

// workspace layout (bytes)
#define WS_W2F   0          // 64*128  f16 B-fragments = 16384 B
#define WS_W3F   16384      // 128*256 f16 B-fragments = 65536 B
#define WS_POOL  81920      // 128*256 f32 pooled      = 131072 B

union AF { v16h v; v8h h[2]; };
union CT { v8f  v; float e[8]; };

// ---------------------------------------------------------------------------
// Prep: convert W2/W3 fp32 -> f16, pre-swizzled into a contiguous-k
// 16x16x32 B-fragment layout (lane = n%16 + 16*(k/16), half = k%16), and
// zero the pooled buffer (relu output >= 0, so 0-init + int atomicMax works).
// ---------------------------------------------------------------------------
__global__ void prep_kernel(const float* __restrict__ W2,
                            const float* __restrict__ W3,
                            char* __restrict__ ws) {
  int i = blockIdx.x * blockDim.x + threadIdx.x;
  _Float16* w2f = (_Float16*)(ws + WS_W2F);
  _Float16* w3f = (_Float16*)(ws + WS_W3F);
  int*      pool = (int*)(ws + WS_POOL);
  if (i < 8192) {                       // W2: KK=2 k-steps, NT=8 n-tiles
    int h = i & 15, L = (i >> 4) & 31, tile = i >> 9;
    int nt = tile & 7, kk = tile >> 3;
    int k = kk * 32 + (L >> 4) * 16 + h;
    int n = nt * 16 + (L & 15);
    w2f[i] = (_Float16)W2[k * C2_ + n];
  } else if (i < 8192 + 32768) {        // W3: KK=4, NT=16
    int j = i - 8192;
    int h = j & 15, L = (j >> 4) & 31, tile = j >> 9;
    int nt = tile & 15, kk = tile >> 4;
    int k = kk * 32 + (L >> 4) * 16 + h;
    int n = nt * 16 + (L & 15);
    w3f[j] = (_Float16)W3[k * C3_ + n];
  } else if (i < 8192 + 32768 + 32768) {
    pool[i - 40960] = 0;                // bits of 0.0f
  }
}

// ---------------------------------------------------------------------------
// Fused point MLP (3->64->128->256, LN+ReLU each) + max-pool over points.
// 256 threads = 8 wave32; each wave owns a 16-point tile per iteration.
// Layers 2/3 via v_wmma_f32_16x16x32_f16; A staged row-major in LDS, read in
// contiguous-k order (k = 16*hi + s); LN stats reduced on C fragments with shfl_xor.
// ---------------------------------------------------------------------------
__global__ __launch_bounds__(256) void mlp_pool_kernel(
    const float* __restrict__ pc,
    const float* __restrict__ W1, const float* __restrict__ b1,
    const float* __restrict__ g1, const float* __restrict__ e1,
    const float* __restrict__ b2, const float* __restrict__ g2, const float* __restrict__ e2,
    const float* __restrict__ b3, const float* __restrict__ g3, const float* __restrict__ e3,
    char* __restrict__ ws)
{
  __shared__ float sW1[3 * 64];
  __shared__ float sB1[64],  sG1[64],  sE1[64];
  __shared__ float sB2[128], sG2[128], sE2[128];
  __shared__ float sB3[256], sG3[256], sE3[256];
  __shared__ __align__(16) _Float16 sA[8][16 * 136];  // per-wave A staging

  const int tid = threadIdx.x;
  for (int i = tid; i < 192; i += 256) sW1[i] = W1[i];
  if (tid < 64)  { sB1[tid] = b1[tid]; sG1[tid] = g1[tid]; sE1[tid] = e1[tid]; }
  if (tid < 128) { sB2[tid] = b2[tid]; sG2[tid] = g2[tid]; sE2[tid] = e2[tid]; }
  sB3[tid] = b3[tid]; sG3[tid] = g3[tid]; sE3[tid] = e3[tid];
  __syncthreads();

  const v16h* W2f  = (const v16h*)(ws + WS_W2F);
  const v16h* W3f  = (const v16h*)(ws + WS_W3F);
  int*        pool = (int*)(ws + WS_POOL);

  const int wave = tid >> 5;
  const int lane = tid & 31;
  const int row  = lane & 15;   // point row within tile / n index in C tiles
  const int hi   = lane >> 4;   // half-wave group (matches WMMA VGPR layout)

  const int CHUNK = 1024;       // points per block
  const int b     = blockIdx.x / (N_ / CHUNK);
  const int chunk = blockIdx.x % (N_ / CHUNK);

  _Float16* A = sA[wave];

  float vmax[16];
  #pragma unroll
  for (int i = 0; i < 16; ++i) vmax[i] = 0.0f;

  #pragma unroll 1
  for (int it = 0; it < CHUNK / 128; ++it) {
    const int p0 = chunk * CHUNK + it * 128 + wave * 16;

    // ---------------- layer 1: 3 -> 64 (scalar, 2 lanes per point) --------
    const float* xp = pc + ((size_t)b * N_ + (p0 + row)) * 3;
    float x0 = xp[0], x1 = xp[1], x2 = xp[2];
    __builtin_prefetch(xp + 128 * 3, 0, 1);   // global_prefetch_b8, next tile

    float hv[32];
    float s = 0.f, s2 = 0.f;
    #pragma unroll
    for (int i = 0; i < 32; ++i) {
      int c = hi * 32 + i;
      float v = x0 * sW1[c] + x1 * sW1[64 + c] + x2 * sW1[128 + c] + sB1[c];
      hv[i] = v; s += v; s2 += v * v;
    }
    s  += __shfl_xor(s, 16, 32);
    s2 += __shfl_xor(s2, 16, 32);
    float mean1 = s * (1.f / 64.f);
    float rstd1 = rsqrtf(s2 * (1.f / 64.f) - mean1 * mean1 + EPS_);
    _Float16 a16[32];
    #pragma unroll
    for (int i = 0; i < 32; ++i) {
      int c = hi * 32 + i;
      float v = (hv[i] - mean1) * rstd1 * sG1[c] + sE1[c];
      a16[i] = (_Float16)fmaxf(v, 0.f);
    }
    #pragma unroll
    for (int j = 0; j < 4; ++j)
      *(v8h*)&A[row * 136 + hi * 32 + j * 8] = *(v8h*)&a16[j * 8];
    asm volatile("s_wait_dscnt 0" ::: "memory");

    // ---------------- layer 2: (16x64)@(64x128) via WMMA ------------------
    AF A0, A1;
    {
      int koff = hi * 16;  // contiguous k = 16*hi + s, matching B fragments
      A0.h[0] = *(const v8h*)&A[row * 136 + 0 * 32 + koff];
      A0.h[1] = *(const v8h*)&A[row * 136 + 0 * 32 + 8 + koff];
      A1.h[0] = *(const v8h*)&A[row * 136 + 1 * 32 + koff];
      A1.h[1] = *(const v8h*)&A[row * 136 + 1 * 32 + 8 + koff];
    }
    CT c2[8];
    float su2[8], sq2[8];
    #pragma unroll
    for (int r = 0; r < 8; ++r) { su2[r] = 0.f; sq2[r] = 0.f; }
    #pragma unroll
    for (int nt = 0; nt < 8; ++nt) {
      CT cc;
      #pragma unroll
      for (int r = 0; r < 8; ++r) cc.e[r] = 0.f;
      cc.v = __builtin_amdgcn_wmma_f32_16x16x32_f16(false, A0.v, false,
              W2f[(0 * 8 + nt) * 32 + lane], (short)0, cc.v, false, false);
      cc.v = __builtin_amdgcn_wmma_f32_16x16x32_f16(false, A1.v, false,
              W2f[(1 * 8 + nt) * 32 + lane], (short)0, cc.v, false, false);
      float bn = sB2[nt * 16 + row];
      #pragma unroll
      for (int r = 0; r < 8; ++r) {
        float v = cc.e[r] + bn; cc.e[r] = v; su2[r] += v; sq2[r] += v * v;
      }
      c2[nt] = cc;
    }
    #pragma unroll
    for (int r = 0; r < 8; ++r) {
      #pragma unroll
      for (int m = 1; m < 16; m <<= 1) {
        su2[r] += __shfl_xor(su2[r], m, 32);
        sq2[r] += __shfl_xor(sq2[r], m, 32);
      }
    }
    #pragma unroll
    for (int nt = 0; nt < 8; ++nt) {
      int n = nt * 16 + row;
      float gg = sG2[n], ee = sE2[n];
      #pragma unroll
      for (int r = 0; r < 8; ++r) {
        float mean = su2[r] * (1.f / 128.f);
        float rstd = rsqrtf(sq2[r] * (1.f / 128.f) - mean * mean + EPS_);
        float v = (c2[nt].e[r] - mean) * rstd * gg + ee;
        A[(r + 8 * hi) * 136 + n] = (_Float16)fmaxf(v, 0.f);
      }
    }
    asm volatile("s_wait_dscnt 0" ::: "memory");

    // ---------------- layer 3: (16x128)@(128x256) via WMMA ----------------
    CT c3[16];
    float su3[8], sq3[8];
    #pragma unroll
    for (int r = 0; r < 8; ++r) { su3[r] = 0.f; sq3[r] = 0.f; }
    #pragma unroll
    for (int nt = 0; nt < 16; ++nt) {
      CT cc;
      #pragma unroll
      for (int r = 0; r < 8; ++r) cc.e[r] = 0.f;
      #pragma unroll
      for (int kk = 0; kk < 4; ++kk) {
        AF af;
        int koff = hi * 16;
        af.h[0] = *(const v8h*)&A[row * 136 + kk * 32 + koff];
        af.h[1] = *(const v8h*)&A[row * 136 + kk * 32 + 8 + koff];
        cc.v = __builtin_amdgcn_wmma_f32_16x16x32_f16(false, af.v, false,
                W3f[(kk * 16 + nt) * 32 + lane], (short)0, cc.v, false, false);
      }
      float bn = sB3[nt * 16 + row];
      #pragma unroll
      for (int r = 0; r < 8; ++r) {
        float v = cc.e[r] + bn; cc.e[r] = v; su3[r] += v; sq3[r] += v * v;
      }
      c3[nt] = cc;
    }
    #pragma unroll
    for (int r = 0; r < 8; ++r) {
      #pragma unroll
      for (int m = 1; m < 16; m <<= 1) {
        su3[r] += __shfl_xor(su3[r], m, 32);
        sq3[r] += __shfl_xor(sq3[r], m, 32);
      }
    }
    #pragma unroll
    for (int nt = 0; nt < 16; ++nt) {
      int n = nt * 16 + row;
      float gg = sG3[n], ee = sE3[n];
      float tm = 0.f;
      #pragma unroll
      for (int r = 0; r < 8; ++r) {
        float mean = su3[r] * (1.f / 256.f);
        float rstd = rsqrtf(sq3[r] * (1.f / 256.f) - mean * mean + EPS_);
        float v = (c3[nt].e[r] - mean) * rstd * gg + ee;
        tm = fmaxf(tm, fmaxf(v, 0.f));          // relu fused into pooling
      }
      tm = fmaxf(tm, __shfl_xor(tm, 16, 32));   // max over both half-waves
      vmax[nt] = fmaxf(vmax[nt], tm);
    }
  }

  // commit pooled maxima (values >= 0 -> int compare is order-preserving)
  if (hi == 0) {
    #pragma unroll
    for (int nt = 0; nt < 16; ++nt)
      atomicMax(&pool[b * C3_ + nt * 16 + row], __float_as_int(vmax[nt]));
  }
}

// ---------------------------------------------------------------------------
// Head: pc_feat = LN(pooled @ Wp + bp); film = pc_feat @ Wf + bf;
// out = film[:,:512] * img + film[:,512:].  Tiny (<0.2% of FLOPs) -> VALU.
// ---------------------------------------------------------------------------
__global__ __launch_bounds__(256) void head_kernel(
    const float* __restrict__ img,
    const float* __restrict__ Wp, const float* __restrict__ bp,
    const float* __restrict__ gp, const float* __restrict__ ep,
    const float* __restrict__ Wf, const float* __restrict__ bfv,
    const char* __restrict__ ws, float* __restrict__ out)
{
  __shared__ float sp[256];
  __shared__ float spc[256];
  __shared__ float sfilm[1024];
  __shared__ float red[2][256];
  const int b = blockIdx.x;
  const int t = threadIdx.x;
  const float* pool = (const float*)(ws + WS_POOL);

  sp[t] = pool[b * 256 + t];
  __syncthreads();

  float acc = bp[t];
  for (int k = 0; k < 256; ++k) acc = fmaf(sp[k], Wp[k * 256 + t], acc);

  red[0][t] = acc; red[1][t] = acc * acc;
  __syncthreads();
  for (int off = 128; off > 0; off >>= 1) {
    if (t < off) { red[0][t] += red[0][t + off]; red[1][t] += red[1][t + off]; }
    __syncthreads();
  }
  float mean = red[0][0] * (1.f / 256.f);
  float rstd = rsqrtf(red[1][0] * (1.f / 256.f) - mean * mean + EPS_);
  spc[t] = (acc - mean) * rstd * gp[t] + ep[t];
  __syncthreads();

  #pragma unroll
  for (int j = 0; j < 4; ++j) {
    int c = t + j * 256;
    float f = bfv[c];
    for (int k = 0; k < 256; ++k) f = fmaf(spc[k], Wf[k * 1024 + c], f);
    sfilm[c] = f;
  }
  __syncthreads();

  #pragma unroll
  for (int j = 0; j < 2; ++j) {
    int i = t + j * 256;
    out[(size_t)b * (2 * IMG_ / 2) * 0 + b * IMG_ + i] =
        sfilm[i] * img[b * IMG_ + i] + sfilm[IMG_ + i];
  }
}

extern "C" void kernel_launch(void* const* d_in, const int* in_sizes, int n_in,
                              void* d_out, int out_size, void* d_ws, size_t ws_size,
                              hipStream_t stream) {
  const float* pc  = (const float*)d_in[0];
  const float* img = (const float*)d_in[1];
  const float* W1  = (const float*)d_in[2];
  const float* b1  = (const float*)d_in[3];
  const float* g1  = (const float*)d_in[4];
  const float* e1  = (const float*)d_in[5];
  const float* W2  = (const float*)d_in[6];
  const float* b2  = (const float*)d_in[7];
  const float* g2  = (const float*)d_in[8];
  const float* e2  = (const float*)d_in[9];
  const float* W3  = (const float*)d_in[10];
  const float* b3  = (const float*)d_in[11];
  const float* g3  = (const float*)d_in[12];
  const float* e3  = (const float*)d_in[13];
  const float* Wp  = (const float*)d_in[14];
  const float* bp  = (const float*)d_in[15];
  const float* gp  = (const float*)d_in[16];
  const float* ep  = (const float*)d_in[17];
  const float* Wf  = (const float*)d_in[18];
  const float* bfv = (const float*)d_in[19];
  char*  ws  = (char*)d_ws;
  float* out = (float*)d_out;

  (void)in_sizes; (void)n_in; (void)out_size; (void)ws_size;

  prep_kernel<<<288, 256, 0, stream>>>(W2, W3, ws);                 // 73728 thr
  mlp_pool_kernel<<<B_ * (N_ / 1024), 256, 0, stream>>>(
      pc, W1, b1, g1, e1, b2, g2, e2, b3, g3, e3, ws);              // 1024 blk
  head_kernel<<<B_, 256, 0, stream>>>(img, Wp, bp, gp, ep, Wf, bfv, ws, out);
}
